// MatchedFilterLoss_34789235098205
// MI455X (gfx1250) — matrix operand order repaired and checked by reading, not compiled, not run on hardware
//
#include <hip/hip_runtime.h>
#include <math.h>

#define N        2048
#define NTH      256
#define MAXSHIFT 51
#define NLAG     103      // 2*51+1
#define NFREQ    993      // masked freqs k = 32..1024
#define KOFF     32
#define NBINS    16

typedef __attribute__((ext_vector_type(16))) _Float16 v16h;
typedef __attribute__((ext_vector_type(8)))  float    v8f;

struct cplx { float x, y; };

// Phase-aliased LDS regions (all members POD so the union is trivially constructible).
union UA {                       // 8 KB
    float hc[N];                 // centered x_hat (phases 1-2)
    float scan[2048];            // [0..1023]=si raw, [1024..2047]=ab raw (phase 3)
};
struct P1 { _Float16 hpad[N + 128]; _Float16 yh[N]; }; // f16 operands for WMMA
union UB {                       // 16 KB
    P1   m;                      // alive only during correlation phase
    cplx z[N];                   // FFT working array (phase 3)
};

__device__ __forceinline__ float block_reduce_sum(float v, float* buf, int tid) {
    buf[tid] = v; __syncthreads();
    for (int s = 128; s > 0; s >>= 1) {
        if (tid < s) buf[tid] = buf[tid] + buf[tid + s];
        __syncthreads();
    }
    float r = buf[0]; __syncthreads();
    return r;
}

// inclusive scan of a[0..1023] using chunk-of-4 + serial chunk-offset pass
__device__ __forceinline__ void block_scan_1024(float* a, float* csum, int tid) {
    int b0 = tid * 4;
    float v0 = a[b0], v1 = a[b0 + 1], v2 = a[b0 + 2], v3 = a[b0 + 3];
    v1 += v0; v2 += v1; v3 += v2;
    csum[tid] = v3;
    __syncthreads();
    if (tid == 0) {
        float run = 0.f;
        for (int i = 0; i < NTH; ++i) { float t = csum[i]; csum[i] = run; run += t; }
    }
    __syncthreads();
    float off = csum[tid];
    a[b0] = v0 + off; a[b0 + 1] = v1 + off; a[b0 + 2] = v2 + off; a[b0 + 3] = v3 + off;
    __syncthreads();
}

__global__ __launch_bounds__(NTH)
void MatchedFilterLoss_kernel(const float* __restrict__ x_hat,
                              const float* __restrict__ x,
                              float* __restrict__ out) {
    __shared__ UA uA;
    __shared__ float yc[N];
    __shared__ UB uB;
    __shared__ float craw[112];
    __shared__ float preS[52], preQ[52], tailS[52], tailQ[52];
    __shared__ float red[NTH];
    __shared__ float red2[NTH];
    __shared__ float csum[NTH];
    __shared__ float sc[4];

    const int tid = threadIdx.x;
    const int row = blockIdx.x;
    const float* xh = x_hat + (size_t)row * N;
    const float* xr = x     + (size_t)row * N;

    // ---- Phase 1: load, center x_hat, center y -------------------------
    float lmin = 3.4e38f, lmax = -3.4e38f, lysum = 0.f;
    for (int i = tid; i < N; i += NTH) {
        float a = xh[i], b = xr[i];
        uA.hc[i] = a; yc[i] = b;
        lmin = fminf(lmin, a); lmax = fmaxf(lmax, a);
        lysum += b;
    }
    red[tid] = lmin; red2[tid] = lmax; __syncthreads();
    for (int s = 128; s > 0; s >>= 1) {
        if (tid < s) {
            red[tid]  = fminf(red[tid],  red[tid + s]);
            red2[tid] = fmaxf(red2[tid], red2[tid + s]);
        }
        __syncthreads();
    }
    float xmin = red[0], xmax = red2[0]; __syncthreads();
    float ymean = block_reduce_sum(lysum, red, tid) * (1.0f / N);

    float scale = 2.0f / (xmax - xmin);
    float lsyy = 0.f, lhs = 0.f, lhq = 0.f;
    for (int i = tid; i < N; i += NTH) {
        float h = (uA.hc[i] - xmin) * scale - 1.0f;
        uA.hc[i] = h;
        float y = yc[i] - ymean;
        yc[i] = y;
        lsyy += y * y; lhs += h; lhq += h * h;
    }
    __syncthreads();
    float Syy  = block_reduce_sum(lsyy, red, tid);
    float Htot = block_reduce_sum(lhs,  red, tid);
    float Qtot = block_reduce_sum(lhq,  red, tid);

    // f16 operands: zero-padded h (64 each side covers +-51 lags), y chunk source
    for (int i = tid; i < N + 128; i += NTH) {
        int s = i - 64;
        uB.m.hpad[i] = (s >= 0 && s < N) ? (_Float16)uA.hc[s] : (_Float16)0.0f;
    }
    for (int i = tid; i < N; i += NTH) uB.m.yh[i] = (_Float16)yc[i];

    // edge prefix sums for per-window mean / variance (windows differ only at edges)
    if (tid < 52) {
        float s0 = 0.f, q0 = 0.f;
        for (int i = 0; i < tid; ++i) { float v = uA.hc[i]; s0 += v; q0 += v * v; }
        preS[tid] = s0; preQ[tid] = q0;
    } else if (tid >= 64 && tid < 116) {
        int n = tid - 64;
        float s0 = 0.f, q0 = 0.f;
        for (int i = N - n; i < N; ++i) { float v = uA.hc[i]; s0 += v; q0 += v * v; }
        tailS[n] = s0; tailQ[n] = q0;
    }
    __syncthreads();

    // ---- Phase 2: Toeplitz cross-correlation via WMMA ------------------
    // A[i][K] = hpad[64 + t0 + K + (lag_base + i)], B[K][n] = yh[t0 + K] (all n)
    {
        int w = tid >> 5, lane = tid & 31;
        int l = lane & 15, hh = lane >> 4;
        int g = (w == 7) ? 6 : w;              // wave 7 redundant: EXEC all-ones for WMMA
        v8f acc = {0.f, 0.f, 0.f, 0.f, 0.f, 0.f, 0.f, 0.f};
        int abase = 13 + g * 16 + l + 8 * hh;  // 64 - 51 = 13
        for (int kt = 0; kt < 64; ++kt) {
            int t0 = kt * 32;
            v16h a, b;
#pragma unroll
            for (int q = 0; q < 16; ++q) {
                // A lane layout: K = q + (q&8) + 8*hh ; B lane layout: K = q + (q&8) + 8*hh
                a[q] = uB.m.hpad[abase + t0 + q + (q & 8)];
                b[q] = uB.m.yh[t0 + q + (q & 8) + 8 * hh];
            }
            acc = __builtin_amdgcn_wmma_f32_16x16x32_f16(
                false, a, false, b, (short)0, acc, false, false);
        }
        if (w != 7 && l == 0) {                // D column 0 lives in lanes 0 (M 0-7) and 16 (M 8-15)
#pragma unroll
            for (int r = 0; r < 8; ++r) craw[g * 16 + r + 8 * hh] = acc[r];
        }
    }
    __syncthreads();

    // ---- corr max -> loss1 --------------------------------------------
    float cval = -3.0f;
    if (tid < NLAG) {
        int s = tid - MAXSHIFT;
        float S, Q;
        if (s >= 0) { S = Htot - preS[s];  Q = Qtot - preQ[s]; }
        else        { S = Htot - tailS[-s]; Q = Qtot - tailQ[-s]; }
        float var = fmaxf(Q - S * S * (1.0f / N), 1e-30f);
        cval = craw[tid] * rsqrtf(var * Syy);
    }
    red[tid] = cval; __syncthreads();
    for (int s = 128; s > 0; s >>= 1) {
        if (tid < s) red[tid] = fmaxf(red[tid], red[tid + s]);
        __syncthreads();
    }
    if (tid == 0) { float cm = red[0]; sc[0] = (1.f - cm) / (1.f + cm); }
    __syncthreads();

    // ---- Phase 3: packed complex FFT (h = re, y = im), in-place DIT ----
    for (int i = tid; i < N; i += NTH) {
        int j = (int)(__brev((unsigned)i) >> 21);   // 11-bit reversal
        uB.z[i].x = uA.hc[j];
        uB.z[i].y = yc[j];
    }
    __syncthreads();
    for (int st = 1; st <= 11; ++st) {
        int len = 1 << st, half = len >> 1;
        float ang = -6.283185307179586f / (float)len;
        for (int idx = tid; idx < N / 2; idx += NTH) {
            int j  = idx & (half - 1);
            int p0 = ((idx >> (st - 1)) << st) + j;
            int p1 = p0 + half;
            float sn, cs; __sincosf(ang * (float)j, &sn, &cs);
            cplx u = uB.z[p0], v = uB.z[p1];
            float tr = v.x * cs - v.y * sn;
            float ti = v.x * sn + v.y * cs;
            uB.z[p0].x = u.x + tr; uB.z[p0].y = u.y + ti;
            uB.z[p1].x = u.x - tr; uB.z[p1].y = u.y - ti;
        }
        __syncthreads();
    }

    // ---- magnitudes (phases cancel in abs(a*conj(b))) -----------------
    const float inv_n = 1.0f / (float)N;            // rfft / SAMPLE_RATE
    for (int m = tid; m < 1024; m += NTH) {
        float a2v = 0.f, abv = 0.f;
        if (m < NFREQ) {
            int k = m + KOFF;
            cplx Zk = uB.z[k];
            cplx Zm = uB.z[(N - k) & (N - 1)];
            float Hre = 0.5f * (Zk.x + Zm.x), Him = 0.5f * (Zk.y - Zm.y);
            float Sre = 0.5f * (Zk.y + Zm.y), Sim = 0.5f * (Zm.x - Zk.x);
            float hm = sqrtf(Hre * Hre + Him * Him) * inv_n;
            float sm = sqrtf(Sre * Sre + Sim * Sim) * inv_n;
            a2v = 4.0f * hm * hm;                   // d(si)
            abv = 4.0f * hm * sm;                   // d(si2) * sqrt(total)
        }
        uA.scan[m]        = a2v;
        uA.scan[1024 + m] = abv;
    }
    __syncthreads();
    block_scan_1024(&uA.scan[0],    csum, tid);
    block_scan_1024(&uA.scan[1024], csum, tid);

    // ---- searchsorted edges, chi-square, final loss -------------------
    if (tid == 0) {
        float total = uA.scan[NFREQ - 1];
        float invs  = rsqrtf(total);
        int edges[NBINS + 1];
        for (int j = 0; j <= NBINS; ++j) {
            float v = total * ((float)j * (1.0f / NBINS));
            int lo = 0, hi = NFREQ;
            while (lo < hi) {                        // searchsorted side='right'
                int mid = (lo + hi) >> 1;
                if (uA.scan[mid] <= v) lo = mid + 1; else hi = mid;
            }
            edges[j] = lo < (NFREQ - 1) ? lo : (NFREQ - 1);
        }
        float total2 = uA.scan[1024 + NFREQ - 1] * invs;
        float tb = total2 * (1.0f / NBINS);
        float accq = 0.f;
        for (int j = 0; j < NBINS; ++j) {
            float r = uA.scan[1024 + edges[j + 1]] * invs;
            float l = uA.scan[1024 + edges[j]]     * invs;
            float d = (r - l) - tb;
            accq += d * d;
        }
        float chisq = (float)NBINS * accq / (float)(NBINS - 1);
        float c3 = chisq * chisq * chisq;
        out[row] = sc[0] * powf(1.0f + c3, 1.0f / 6.0f);
    }
}

extern "C" void kernel_launch(void* const* d_in, const int* in_sizes, int n_in,
                              void* d_out, int out_size, void* d_ws, size_t ws_size,
                              hipStream_t stream) {
    (void)n_in; (void)out_size; (void)d_ws; (void)ws_size;
    const float* x_hat = (const float*)d_in[0];
    const float* x     = (const float*)d_in[1];
    float* out = (float*)d_out;
    int rows = in_sizes[0] / N;      // 256*2 = 512 rows, one block each
    MatchedFilterLoss_kernel<<<dim3(rows), dim3(NTH), 0, stream>>>(x_hat, x, out);
}
